// SpaceTimeTransformer_2422361555421
// MI455X (gfx1250) — matrix operand order repaired and checked by reading, compiled, not run on hardware
//
#include <hip/hip_runtime.h>
#include <hip/hip_bf16.h>

typedef __attribute__((ext_vector_type(16))) _Float16 v16h;
typedef __attribute__((ext_vector_type(8)))  float    v8f;
typedef __attribute__((ext_vector_type(8)))  unsigned int v8u;

#define D_MODEL 256
#define NHEAD   8
#define DKH     32
#define FF_DIM  1024
#define BB      2
#define TT      16
#define NNODES  675
#define KEEP_N  540
#define NTOK    (BB*NNODES*TT)   /* 21600 */
#define KTOK    (BB*KEEP_N*TT)   /* 17280 */

__device__ __forceinline__ v8f wmma16(v16h a, v16h b, v8f c) {
  return __builtin_amdgcn_wmma_f32_16x16x32_f16(false, a, false, b, (short)0, c,
                                                false, false);
}

// ---- DPP16 butterfly reductions (no LDS, no wait counters) ---------------
// 0xB1 quad_perm(1,0,3,2)=xor1, 0x4E quad_perm(2,3,0,1)=xor2,
// 0x141 row_half_mirror (xor7), 0x140 row_mirror (xor15).
template <int CTRL>
__device__ __forceinline__ float dpp_mov(float x) {
  return __builtin_bit_cast(float,
      __builtin_amdgcn_update_dpp(0, __builtin_bit_cast(int, x), CTRL, 0xf, 0xf, true));
}
__device__ __forceinline__ float red_max16(float x) {
  x = fmaxf(x, dpp_mov<0xB1>(x));
  x = fmaxf(x, dpp_mov<0x4E>(x));
  x = fmaxf(x, dpp_mov<0x141>(x));
  x = fmaxf(x, dpp_mov<0x140>(x));
  return x;
}
__device__ __forceinline__ float red_sum16(float x) {
  x += dpp_mov<0xB1>(x);
  x += dpp_mov<0x4E>(x);
  x += dpp_mov<0x141>(x);
  x += dpp_mov<0x140>(x);
  return x;
}
__device__ __forceinline__ float red_sum32(float x) {
  x = red_sum16(x);
  x += __shfl_xor(x, 16, 32);
  return x;
}

// ---------------------------------------------------------------------------
// Weight pre-swizzle: W[K,N] f32 -> f16 blocks in WMMA B-fragment order.
// ---------------------------------------------------------------------------
__global__ __launch_bounds__(128)
void wconv(const float* __restrict__ W, _Float16* __restrict__ WH, int K, int N)
{
  const int wave = threadIdx.x >> 5, lane = threadIdx.x & 31;
  const int half = lane >> 4, lm = lane & 15;
  const int n16 = N >> 4;
  const int blk = blockIdx.x * 4 + wave;
  if (blk >= (K >> 5) * n16) return;
  const int kb = blk / n16, nt = blk % n16;
  const int col = nt * 16 + lm;
  _Float16* dst = WH + ((size_t)blk * 32 + lane) * 16;
#pragma unroll
  for (int j = 0; j < 8; ++j) {
    const int k = kb * 32 + half * 16 + 2 * j;
    dst[2 * j]     = (_Float16)W[(size_t)k * N + col];
    dst[2 * j + 1] = (_Float16)W[(size_t)(k + 1) * N + col];
  }
}

// ---------------------------------------------------------------------------
// GEMM: act(A[M,K] * W[K,N] + bias) * oscale.  One wave -> 16x64 tile.
// ACT:  0 none, 1 relu, 2 gelu(exact)
// MODE: 0 f32 C; 1 f16 Ch; 2 f16 V space-transposed; 3 f16 V time-transposed
// ---------------------------------------------------------------------------
template <int ACT, int MODE>
__global__ __launch_bounds__(128)
void gemm_wmma(const float* __restrict__ A, const v16h* __restrict__ WH,
               const float* __restrict__ bias, float* __restrict__ Cf,
               _Float16* __restrict__ Ch, int M, int K, int N,
               int dNK, int dPad, float oscale)
{
  const int wave = threadIdx.x >> 5, lane = threadIdx.x & 31;
  const int half = lane >> 4, lm = lane & 15;
  const int ngrp = N >> 6;
  const int total = (M >> 4) * ngrp;
  const int tile = blockIdx.x * 4 + wave;
  if (tile >= total) return;
  const int mt = tile / ngrp, ng = tile % ngrp;
  const int n16 = N >> 4;
  const float* __restrict__ arow = A + (size_t)(mt * 16 + lm) * K;

  v8f acc[4] = {};
  int kb = 0;
  for (int k0 = 0; k0 < K; k0 += 32, ++kb) {
    __builtin_prefetch(arow + k0 + 64, 0, 0);
    v16h a;
#pragma unroll
    for (int p = 0; p < 8; ++p) {
      const int ka = k0 + half * 16 + 2 * p;
      const float2 f = *(const float2*)(arow + ka);
      a[2 * p] = (_Float16)f.x;
      a[2 * p + 1] = (_Float16)f.y;
    }
    const v16h* wb = WH + ((size_t)kb * n16 + ng * 4) * 32 + lane;
    acc[0] = wmma16(a, wb[0],  acc[0]);
    acc[1] = wmma16(a, wb[32], acc[1]);
    acc[2] = wmma16(a, wb[64], acc[2]);
    acc[3] = wmma16(a, wb[96], acc[3]);
  }

#pragma unroll
  for (int s = 0; s < 4; ++s) {
    const int col = (ng * 4 + s) * 16 + lm;
    const float bv = bias[col];
#pragma unroll
    for (int i = 0; i < 8; ++i) {
      const int r = mt * 16 + i + 8 * half;
      float v = acc[s][i] + bv;
      if (ACT == 1) v = fmaxf(v, 0.0f);
      if (ACT == 2) v = 0.5f * v * (1.0f + erff(v * 0.70710678118654752f));
      v *= oscale;
      if (MODE == 0) {
        Cf[(size_t)r * N + col] = v;
      } else if (MODE == 1) {
        Ch[(size_t)r * N + col] = (_Float16)v;
      } else if (MODE == 2) {
        const int b = r / (dNK * TT);
        const int key = (r / TT) % dNK;
        const int t = r % TT;
        Ch[((size_t)((b * TT + t) * D_MODEL) + col) * dPad + key] = (_Float16)v;
      } else {
        const int rb = r / TT;
        const int t = r % TT;
        Ch[((size_t)rb * D_MODEL + col) * TT + t] = (_Float16)v;
      }
    }
  }
}

// Zero the pad column(s) of the space-transposed V when NK is odd.
__global__ __launch_bounds__(256)
void pad_vth(_Float16* __restrict__ VTH, int NK, int NKpad, int rows)
{
  const int pad = NKpad - NK;
  const int i = blockIdx.x * 256 + threadIdx.x;
  if (i >= rows * pad) return;
  const int r = i / pad, o = i % pad;
  VTH[(size_t)r * NKpad + NK + o] = (_Float16)0.0f;
}

// ---------------------------------------------------------------------------
// Space attention (flash), all-f16 operands, DPP softmax reductions.
// ---------------------------------------------------------------------------
__global__ __launch_bounds__(32)
void attn_space(const _Float16* __restrict__ QH, const _Float16* __restrict__ KH,
                const _Float16* __restrict__ VTH, float* __restrict__ O,
                int NK, int NKpad)
{
  __shared__ _Float16 pl[16 * 32];
  const int lane = threadIdx.x;
  const int half = lane >> 4, lm = lane & 15;
  const int nt = blockIdx.x;
  const int b = blockIdx.y / TT, t = blockIdx.y % TT;
  const int h = blockIdx.z;

  int qn = nt * 16 + lm; if (qn > NNODES - 1) qn = NNODES - 1;
  const _Float16* qrow = QH + ((size_t)((b * NNODES + qn) * TT + t)) * D_MODEL + h * DKH;
  v8u au;
#pragma unroll
  for (int p = 0; p < 8; ++p) {
    const int k = half * 16 + 2 * p;
    au[p] = *(const unsigned int*)(qrow + k);
  }
  const v16h aq = __builtin_bit_cast(v16h, au);

  float mrow[8], lrow[8];
#pragma unroll
  for (int i = 0; i < 8; ++i) { mrow[i] = -1e30f; lrow[i] = 0.0f; }
  v8f o0 = {}, o1 = {};
  const _Float16* vt0 = VTH + ((size_t)((b * TT + t) * D_MODEL) + h * DKH + lm) * NKpad;
  const _Float16* vt1 = vt0 + (size_t)16 * NKpad;

  for (int m0 = 0; m0 < NK; m0 += 32) {
    v8u ku0, ku1;
    {
      int key = m0 + lm; if (key > NK - 1) key = NK - 1;
      const _Float16* kr = KH + ((size_t)((b * NK + key) * TT + t)) * D_MODEL + h * DKH;
#pragma unroll
      for (int j = 0; j < 8; ++j)
        ku0[j] = *(const unsigned int*)(kr + half * 16 + 2 * j);
      key = m0 + 16 + lm; if (key > NK - 1) key = NK - 1;
      kr = KH + ((size_t)((b * NK + key) * TT + t)) * D_MODEL + h * DKH;
#pragma unroll
      for (int j = 0; j < 8; ++j)
        ku1[j] = *(const unsigned int*)(kr + half * 16 + 2 * j);
    }
    v8f z = {};
    v8f s0 = wmma16(aq, __builtin_bit_cast(v16h, ku0), z);
    v8f s1 = wmma16(aq, __builtin_bit_cast(v16h, ku1), z);
    const bool inv0 = (m0 + lm) >= NK;
    const bool inv1 = (m0 + 16 + lm) >= NK;
#pragma unroll
    for (int i = 0; i < 8; ++i) {
      if (inv0) s0[i] = -1e30f;
      if (inv1) s1[i] = -1e30f;
    }
#pragma unroll
    for (int i = 0; i < 8; ++i) {
      const float mx = red_max16(fmaxf(s0[i], s1[i]));
      const float mnew = fmaxf(mrow[i], mx);
      const float sc = expf(mrow[i] - mnew);
      const float p0 = expf(s0[i] - mnew);
      const float p1 = expf(s1[i] - mnew);
      const float rs = red_sum16(p0 + p1);
      lrow[i] = lrow[i] * sc + rs;
      mrow[i] = mnew;
      o0[i] *= sc; o1[i] *= sc;
      pl[(i + 8 * half) * 32 + lm]      = (_Float16)p0;
      pl[(i + 8 * half) * 32 + 16 + lm] = (_Float16)p1;
    }
    __syncthreads();
    v8u pu;
#pragma unroll
    for (int p = 0; p < 8; ++p) {
      const int k = half * 16 + 2 * p;
      pu[p] = *(const unsigned int*)&pl[lm * 32 + k];
    }
    const v16h ap = __builtin_bit_cast(v16h, pu);
    v8u vu0, vu1;
#pragma unroll
    for (int j = 0; j < 8; ++j) {
      int kk = m0 + half * 16 + 2 * j;
      if (kk > NKpad - 2) kk = NKpad - 2;     // stay inside finite, written row
      vu0[j] = *(const unsigned int*)(vt0 + kk);
      vu1[j] = *(const unsigned int*)(vt1 + kk);
    }
    o0 = wmma16(ap, __builtin_bit_cast(v16h, vu0), o0);
    o1 = wmma16(ap, __builtin_bit_cast(v16h, vu1), o1);
    __syncthreads();
  }

#pragma unroll
  for (int i = 0; i < 8; ++i) {
    const int n = nt * 16 + i + 8 * half;
    if (n < NNODES) {
      const float inv = 1.0f / lrow[i];
      float* orow = O + ((size_t)((b * NNODES + n) * TT + t)) * D_MODEL + h * DKH;
      orow[lm]      = o0[i] * inv;
      orow[16 + lm] = o1[i] * inv;
    }
  }
}

// ---------------------------------------------------------------------------
// Time attention: per (b,h,n), T=16.  VTH: ((b,n,ch),t) f16.
// ---------------------------------------------------------------------------
__global__ __launch_bounds__(32)
void attn_time(const _Float16* __restrict__ QH, const _Float16* __restrict__ KH,
               const _Float16* __restrict__ VTH, float* __restrict__ O)
{
  __shared__ _Float16 pl[16 * 32];
  const int lane = threadIdx.x;
  const int half = lane >> 4, lm = lane & 15;
  const int n = blockIdx.x, b = blockIdx.y, h = blockIdx.z;
  const size_t tok0 = (size_t)(b * NNODES + n) * TT;

  v8u au, ku;
  const _Float16* qrow = QH + (tok0 + lm) * D_MODEL + h * DKH;
  const _Float16* krow = KH + (tok0 + lm) * D_MODEL + h * DKH;
#pragma unroll
  for (int p = 0; p < 8; ++p) {
    const int k = half * 16 + 2 * p;
    au[p] = *(const unsigned int*)(qrow + k);
  }
#pragma unroll
  for (int j = 0; j < 8; ++j)
    ku[j] = *(const unsigned int*)(krow + half * 16 + 2 * j);

  v8f z = {};
  v8f s = wmma16(__builtin_bit_cast(v16h, au), __builtin_bit_cast(v16h, ku), z);
#pragma unroll
  for (int i = 0; i < 8; ++i) {
    const float mx = red_max16(s[i]);
    const float e = expf(s[i] - mx);
    const float rs = red_sum16(e);
    pl[(i + 8 * half) * 32 + lm]      = (_Float16)(e / rs);
    pl[(i + 8 * half) * 32 + 16 + lm] = (_Float16)0.0f;  // pad K 16..31
  }
  __syncthreads();
  v8u pu;
#pragma unroll
  for (int p = 0; p < 8; ++p) {
    const int k = half * 16 + 2 * p;
    pu[p] = *(const unsigned int*)&pl[lm * 32 + k];
  }
  const v16h ap = __builtin_bit_cast(v16h, pu);

  const _Float16* vt0 = VTH + ((size_t)(b * NNODES + n) * D_MODEL + h * DKH + lm) * TT;
  const _Float16* vt1 = vt0 + (size_t)16 * TT;
  v8u vu0, vu1;
#pragma unroll
  for (int j = 0; j < 8; ++j) {
    const int kk = (half * 16 + 2 * j) & 15;  // K>=16 rows hit P==0 anyway
    vu0[j] = *(const unsigned int*)(vt0 + kk);
    vu1[j] = *(const unsigned int*)(vt1 + kk);
  }
  v8f o0 = wmma16(ap, __builtin_bit_cast(v16h, vu0), z);
  v8f o1 = wmma16(ap, __builtin_bit_cast(v16h, vu1), z);
#pragma unroll
  for (int i = 0; i < 8; ++i) {
    float* orow = O + (tok0 + i + 8 * half) * D_MODEL + h * DKH;
    orow[lm]      = o0[i];
    orow[16 + lm] = o1[i];
  }
}

// ---------------------------------------------------------------------------
// Y = LayerNorm(X [+ A]) over D=256.  One wave per token row.
// ---------------------------------------------------------------------------
__global__ __launch_bounds__(256)
void add_layernorm(const float* __restrict__ X, const float* __restrict__ Aa,
                   const float* __restrict__ g, const float* __restrict__ be,
                   float* __restrict__ Y, int M, int do_add)
{
  const int wave = threadIdx.x >> 5, lane = threadIdx.x & 31;
  const int row = blockIdx.x * 8 + wave;
  if (row >= M) return;
  const float* xr = X  + (size_t)row * D_MODEL;
  const float* ar = Aa + (size_t)row * D_MODEL;
  float v[8], s = 0.0f, sq = 0.0f;
#pragma unroll
  for (int i = 0; i < 8; ++i) {
    const int c = lane + 32 * i;
    float tv = xr[c];
    if (do_add) tv += ar[c];
    v[i] = tv; s += tv; sq += tv * tv;
  }
  s  = red_sum32(s);
  sq = red_sum32(sq);
  const float mean = s * (1.0f / 256.0f);
  const float var  = sq * (1.0f / 256.0f) - mean * mean;
  const float inv  = rsqrtf(var + 1e-5f);
  float* yr = Y + (size_t)row * D_MODEL;
#pragma unroll
  for (int i = 0; i < 8; ++i) {
    const int c = lane + 32 * i;
    yr[c] = (v[i] - mean) * inv * g[c] + be[c];
  }
}

// ---------------------------------------------------------------------------
// Predictor helpers
// ---------------------------------------------------------------------------
__global__ __launch_bounds__(128)
void pred_glob(const float* __restrict__ H, float* __restrict__ G)
{
  const int bt = blockIdx.x, b = bt / TT, t = bt % TT;
  const int d2 = threadIdx.x;
  float s = 0.0f;
  for (int n = 0; n < NNODES; ++n)
    s += H[((size_t)((b * NNODES + n) * TT + t)) * D_MODEL + 128 + d2];
  G[bt * 128 + d2] = s * (1.0f / (float)NNODES);
}

__global__ __launch_bounds__(256)
void pred_concat(const float* __restrict__ H, const float* __restrict__ G,
                 float* __restrict__ Z)
{
  const int row = blockIdx.x;
  const int c = threadIdx.x;
  const int b = row / (NNODES * TT);
  const int t = (row % (NNODES * TT)) % TT;
  float v;
  if (c < 128) v = H[(size_t)row * D_MODEL + c];
  else         v = G[(b * TT + t) * 128 + (c - 128)];
  Z[(size_t)row * D_MODEL + c] = v;
}

__global__ __launch_bounds__(256)
void pred_score(const float* __restrict__ Z2, const float* __restrict__ w3,
                const float* __restrict__ b3, float* __restrict__ SC)
{
  const int wave = threadIdx.x >> 5, lane = threadIdx.x & 31;
  const int row = blockIdx.x * 8 + wave;
  if (row >= NTOK) return;
  const float* zr = Z2 + (size_t)row * 64;
  float a0 = 0.0f, a1 = 0.0f;
#pragma unroll
  for (int i = 0; i < 2; ++i) {
    const int k = lane + 32 * i;
    const float zv = zr[k];
    a0 += zv * w3[k * 2];
    a1 += zv * w3[k * 2 + 1];
  }
  a0 = red_sum32(a0);
  a1 = red_sum32(a1);
  if (lane == 0) {
    a0 += b3[0]; a1 += b3[1];
    const float mx = fmaxf(a0, a1);
    const float ls = mx + logf(expf(a0 - mx) + expf(a1 - mx));
    const int b = row / (NNODES * TT);
    const int rem = row % (NNODES * TT);
    const int n = rem / TT, t = rem % TT;
    SC[(b * TT + t) * NNODES + n] = a0 - ls;
  }
}

__global__ __launch_bounds__(256)
void topk_keep(const float* __restrict__ SC, int* __restrict__ IDX)
{
  __shared__ float s[NNODES];
  __shared__ unsigned char keepf[NNODES];
  const int bt = blockIdx.x;
  for (int i = threadIdx.x; i < NNODES; i += 256) s[i] = SC[bt * NNODES + i];
  __syncthreads();
  for (int i = threadIdx.x; i < NNODES; i += 256) {
    const float si = s[i];
    int rank = 0;
    for (int j = 0; j < NNODES; ++j) {
      const float sj = s[j];
      rank += (sj > si) || ((sj == si) && (j < i));
    }
    keepf[i] = (rank < KEEP_N) ? 1 : 0;
  }
  __syncthreads();
  if (threadIdx.x == 0) {
    int pos = 0;
    for (int i = 0; i < NNODES; ++i)
      if (keepf[i]) IDX[bt * KEEP_N + (pos++)] = i;
  }
}

__global__ __launch_bounds__(256)
void gather_kv(const float* __restrict__ X, const int* __restrict__ IDX,
               float* __restrict__ KB)
{
  const int bj = blockIdx.x, b = bj / KEEP_N, j = bj % KEEP_N;
  const int t = blockIdx.y, c = threadIdx.x;
  const int n = IDX[(b * TT + t) * KEEP_N + j];
  KB[((size_t)((b * KEEP_N + j) * TT + t)) * D_MODEL + c] =
      X[((size_t)((b * NNODES + n) * TT + t)) * D_MODEL + c];
}

// ---------------------------------------------------------------------------
// Host orchestration
// ---------------------------------------------------------------------------
struct MhaP { const float *qw,*qb,*kw,*kb,*vw,*vb,*ow,*ob; };

extern "C" void kernel_launch(void* const* d_in, const int* in_sizes, int n_in,
                              void* d_out, int out_size, void* d_ws, size_t ws_size,
                              hipStream_t stream)
{
  (void)in_sizes; (void)n_in; (void)out_size; (void)ws_size;
  auto F = [&](int i) { return (const float*)d_in[i]; };

  const float* x_in = F(0);
  MhaP sp { F(1),F(2),F(3),F(4),F(5),F(6),F(7),F(8) };
  MhaP tm { F(9),F(10),F(11),F(12),F(13),F(14),F(15),F(16) };
  const float *ln0g=F(17), *ln0b=F(18), *ln1g=F(19), *ln1b=F(20);
  const float *f1w=F(21), *f1b=F(22), *f2w=F(23), *f2b=F(24);
  const float *plg=F(25), *plb=F(26), *pinw=F(27), *pinb=F(28);
  const float *c1w=F(29), *c1b=F(30), *c2w=F(31), *c2b=F(32);
  const float *c3w=F(33), *c3b=F(34);

  const size_t TOKBUF = (size_t)NTOK * D_MODEL;
  float* ws = (float*)d_ws;
  float* XA = ws;
  float* XB = XA + TOKBUF;
  float* XL = XB + TOKBUF;
  float* T1 = XL + TOKBUF;
  float* Q  = T1 + TOKBUF;
  float* K  = Q  + TOKBUF;
  float* V  = K  + TOKBUF;
  float* T2 = V  + TOKBUF;
  float* H1 = Q;                      // FFN hidden aliases Q..T2 (dead then)
  float* KB = T2 + TOKBUF;
  float* SC = KB + (size_t)KTOK * D_MODEL;
  float* GL = SC + (size_t)BB * TT * NNODES;
  int*  IDX = (int*)(GL + (size_t)BB * TT * 128);
  _Float16* WHB = (_Float16*)(IDX + BB * TT * KEEP_N + 64);

  _Float16* QH  = (_Float16*)Q;       // f16 attention operands alias Q/K/V
  _Float16* KH  = (_Float16*)K;
  _Float16* VTH = (_Float16*)V;

  // ---- pre-swizzle all GEMM weights to f16 fragment layout --------------
  size_t woff = 0;
  auto conv = [&](const float* W, int Kd, int N) {
    _Float16* dst = WHB + woff;
    const int blocks = (Kd >> 5) * (N >> 4);
    wconv<<<(blocks + 3) / 4, 128, 0, stream>>>(W, dst, Kd, N);
    woff += (size_t)Kd * N;
    return (const _Float16*)dst;
  };
  const _Float16* w_spq = conv(sp.qw, 256, 256);
  const _Float16* w_spk = conv(sp.kw, 256, 256);
  const _Float16* w_spv = conv(sp.vw, 256, 256);
  const _Float16* w_spo = conv(sp.ow, 256, 256);
  const _Float16* w_tmq = conv(tm.qw, 256, 256);
  const _Float16* w_tmk = conv(tm.kw, 256, 256);
  const _Float16* w_tmv = conv(tm.vw, 256, 256);
  const _Float16* w_tmo = conv(tm.ow, 256, 256);
  const _Float16* w_f1  = conv(f1w, 256, 1024);
  const _Float16* w_f2  = conv(f2w, 1024, 256);
  const _Float16* w_pin = conv(pinw, 256, 256);
  const _Float16* w_c1  = conv(c1w, 256, 128);
  const _Float16* w_c2  = conv(c2w, 128, 64);

  auto launch = [&](auto kern, const float* A, const _Float16* W,
                    const float* bias, float* Cf, _Float16* Ch,
                    int M, int Kd, int N, int dNK, int dPad, float osc) {
    const int tiles = (M / 16) * (N / 64);
    kern<<<(tiles + 3) / 4, 128, 0, stream>>>(
        A, (const v16h*)W, bias, Cf, Ch, M, Kd, N, dNK, dPad, osc);
  };

  const float SCALE = 0.17677669529663687f;  // 1/sqrt(32)

  auto enc = [&](const float* Xin, const float* KV, int NK, float* Xout) {
    const int Mq = NTOK;
    const int Mk = BB * NK * TT;
    const int NKpad = (NK + 1) & ~1;
    // space MHA
    launch(gemm_wmma<0,1>, Xin, w_spq, sp.qb, nullptr, QH,  Mq, 256, 256, 0, 0, SCALE);
    launch(gemm_wmma<0,1>, KV,  w_spk, sp.kb, nullptr, KH,  Mk, 256, 256, 0, 0, 1.0f);
    launch(gemm_wmma<0,2>, KV,  w_spv, sp.vb, nullptr, VTH, Mk, 256, 256, NK, NKpad, 1.0f);
    if (NKpad > NK) {
      const int pe = BB * TT * D_MODEL * (NKpad - NK);
      pad_vth<<<(pe + 255) / 256, 256, 0, stream>>>(VTH, NK, NKpad, BB * TT * D_MODEL);
    }
    attn_space<<<dim3((NNODES + 15) / 16, BB * TT, NHEAD), 32, 0, stream>>>(
        QH, KH, VTH, T1, NK, NKpad);
    launch(gemm_wmma<0,0>, T1, w_spo, sp.ob, T2, nullptr, Mq, 256, 256, 0, 0, 1.0f);
    // time MHA
    launch(gemm_wmma<0,1>, T2, w_tmq, tm.qb, nullptr, QH,  Mq, 256, 256, 0, 0, SCALE);
    launch(gemm_wmma<0,1>, T2, w_tmk, tm.kb, nullptr, KH,  Mq, 256, 256, 0, 0, 1.0f);
    launch(gemm_wmma<0,3>, T2, w_tmv, tm.vb, nullptr, VTH, Mq, 256, 256, 0, 16, 1.0f);
    attn_time<<<dim3(NNODES, BB, NHEAD), 32, 0, stream>>>(QH, KH, VTH, T1);
    launch(gemm_wmma<0,0>, T1, w_tmo, tm.ob, T2, nullptr, Mq, 256, 256, 0, 0, 1.0f);
    // residual + LN + FFN + LN
    add_layernorm<<<(Mq + 7) / 8, 256, 0, stream>>>(Xin, T2, ln0g, ln0b, XL, Mq, 1);
    launch(gemm_wmma<1,0>, XL, w_f1, f1b, H1, nullptr, Mq, 256, FF_DIM, 0, 0, 1.0f);
    launch(gemm_wmma<0,0>, H1, w_f2, f2b, T1, nullptr, Mq, FF_DIM, 256, 0, 0, 1.0f);
    add_layernorm<<<(Mq + 7) / 8, 256, 0, stream>>>(XL, T1, ln1g, ln1b, Xout, Mq, 1);
  };

  // Layer 0
  enc(x_in, x_in, NNODES, XA);

  // Predictor + top-k gather (PRUNE_LOC == 1)
  add_layernorm<<<(NTOK + 7) / 8, 256, 0, stream>>>(XA, XA, plg, plb, T1, NTOK, 0);
  launch(gemm_wmma<2,0>, T1, w_pin, pinb, Q, nullptr, NTOK, 256, 256, 0, 0, 1.0f);
  pred_glob<<<BB * TT, 128, 0, stream>>>(Q, GL);
  pred_concat<<<NTOK, 256, 0, stream>>>(Q, GL, K);
  launch(gemm_wmma<2,0>, K, w_c1, c1b, V, nullptr, NTOK, 256, 128, 0, 0, 1.0f);
  launch(gemm_wmma<2,0>, V, w_c2, c2b, T2, nullptr, NTOK, 128, 64, 0, 0, 1.0f);
  pred_score<<<(NTOK + 7) / 8, 256, 0, stream>>>(T2, c3w, c3b, SC);
  topk_keep<<<BB * TT, 256, 0, stream>>>(SC, IDX);
  gather_kv<<<dim3(BB * KEEP_N, TT), 256, 0, stream>>>(XA, IDX, KB);

  // Layer 1 (pruned keys/values)
  enc(XA, KB, KEEP_N, XB);

  // Layer 2
  enc(XB, XB, NNODES, (float*)d_out);
}
